// Encoder_37091337568879
// MI455X (gfx1250) — compile-verified, matrix operand order checked by reading, not run on hardware
//
#include <hip/hip_runtime.h>

typedef __attribute__((ext_vector_type(16))) _Float16 v16h;
typedef __attribute__((ext_vector_type(8)))  _Float16 v8h;
typedef __attribute__((ext_vector_type(8)))  float    v8f;
typedef __attribute__((ext_vector_type(4)))  float    v4f;

#define VOCAB 50000
#define DDIM  128
#define TLEN  256
#define BSZ   512
#define HDIM  64
#define NH4   256   // 4*H

// Workspace layout (bytes):
//   [0, 65536)               WxT  f16 [256][128]   (Wx transposed: WxT[n][k] = Wx[k][n])
//   [65536, 98304)           WhT  f16 [256][64]
//   [98304, 98304+134217728) xz   f32, fragment-major: ((b_tile*T + t)*16 + nt)*256 + lane*8
#define WS_WHT_ELEMS (NH4*DDIM)                   // f16 element offset of WhT
#define WS_XZ_BYTES  (NH4*DDIM*2 + NH4*HDIM*2)    // 98304, 256B aligned

static __device__ inline v16h combine16(v8h lo, v8h hi) {
    v16h r;
#pragma unroll
    for (int e = 0; e < 8; ++e) { r[e] = lo[e]; r[e + 8] = hi[e]; }
    return r;
}

static __device__ inline float sigmoidf_fast(float x) {
    return 1.0f / (1.0f + __expf(-x));
}
static __device__ inline float tanhf_fast(float x) {
    float e = __expf(-2.0f * x);
    return (1.0f - e) / (1.0f + e);
}

// ---------------------------------------------------------------------------
// Kernel 0: convert + transpose weights to f16 for WMMA B-fragment loading.
// ---------------------------------------------------------------------------
__global__ __launch_bounds__(256) void prep_weights(const float* __restrict__ Wx,
                                                    const float* __restrict__ Wh,
                                                    _Float16* __restrict__ wxt,
                                                    _Float16* __restrict__ wht) {
    int i = blockIdx.x * blockDim.x + threadIdx.x;
    if (i < NH4 * DDIM) {
        int n = i / DDIM, k = i % DDIM;
        wxt[i] = (_Float16)Wx[k * NH4 + n];
    }
    int j = i - NH4 * DDIM;
    if (j >= 0 && j < NH4 * HDIM) {
        int n = j / HDIM, k = j % HDIM;
        wht[j] = (_Float16)Wh[k * NH4 + n];
    }
}

// ---------------------------------------------------------------------------
// Kernel 1: xz = gather(emb, tokens) @ Wx + b, stored fragment-major in ws.
// 256 workgroups x 4 waves; each workgroup async-stages the full WxT (64KB
// f16) into padded LDS once (global_load_async_to_lds_b128, ASYNCcnt), then
// each wave runs 8 (b_tile,t) tiles: M=16, K=128, N=256 -> 64 WMMAs/tile with
// conflict-free ds_load_b128 B-fragments.
// ---------------------------------------------------------------------------
__global__ __launch_bounds__(128) void xz_gemm(const int* __restrict__ tokens,
                                               const float* __restrict__ emb,
                                               const _Float16* __restrict__ wxt,
                                               const float* __restrict__ bias,
                                               float* __restrict__ xz) {
    const int lane  = threadIdx.x & 31;
    const int wave  = threadIdx.x >> 5;
    const int m     = lane & 15;               // row within tile / N within tile
    const int khalf = (lane >> 4) * 8;         // upper half-wave holds K+8 block
    const int WXP   = 136;                     // padded LDS pitch (f16): 272B -> conflict-free

    __shared__ __align__(16) _Float16 wx_lds[NH4 * 136];

    // ---- Stage WxT -> LDS with CDNA5 async copies (no VGPR roundtrip). ----
    {
        const char* gbase = (const char*)wxt;
        unsigned    lbase = (unsigned)(unsigned long long)(&wx_lds[0]);
        for (int c = threadIdx.x; c < NH4 * 16; c += 128) {   // 4096 x 16B chunks
            int row = c >> 4;                  // n in [0,256)
            int cb  = (c & 15) * 16;           // byte offset within 256B row
            unsigned    laddr = lbase + (unsigned)(row * (WXP * 2) + cb);
            const char* gaddr = gbase + row * 256 + cb;
            asm volatile("global_load_async_to_lds_b128 %0, %1, off"
                         :: "v"(laddr), "v"(gaddr) : "memory");
        }
        asm volatile("s_wait_asynccnt 0x0" ::: "memory");
    }
    __syncthreads();

    for (int it = 0; it < 8; ++it) {
        const int gw = (blockIdx.x * 4 + wave) + it * 1024;  // 0..8191 = b_tile*256 + t
        const int bt = gw >> 8;
        const int t  = gw & 255;

        // Gather one embedding row per lane-row into A fragments (f32 -> f16).
        const int tok = tokens[(bt * 16 + m) * TLEN + t];
        const float* erow = emb + (long)tok * DDIM;

        v16h a[4];
#pragma unroll
        for (int kt = 0; kt < 4; ++kt) {
            const float* p = erow + kt * 32 + khalf;
            v4f x0 = *(const v4f*)(p);
            v4f x1 = *(const v4f*)(p + 4);
            v4f x2 = *(const v4f*)(p + 16);
            v4f x3 = *(const v4f*)(p + 20);
            v16h av;
#pragma unroll
            for (int e = 0; e < 4; ++e) {
                av[e]      = (_Float16)x0[e];
                av[e + 4]  = (_Float16)x1[e];
                av[e + 8]  = (_Float16)x2[e];
                av[e + 12] = (_Float16)x3[e];
            }
            a[kt] = av;
        }

#pragma unroll 4
        for (int nt = 0; nt < 16; ++nt) {
            float bb = bias[nt * 16 + m];
            v8f acc = { bb, bb, bb, bb, bb, bb, bb, bb };
#pragma unroll
            for (int kt = 0; kt < 4; ++kt) {
                const _Float16* bp = &wx_lds[(nt * 16 + m) * WXP + kt * 32 + khalf];
                v8h lo = *(const v8h*)bp;
                v8h hi = *(const v8h*)(bp + 16);
                v16h bf = combine16(lo, hi);
                acc = __builtin_amdgcn_wmma_f32_16x16x32_f16(
                    false, a[kt], false, bf, (short)0, acc, false, false);
            }
            // Fragment-major store: scan kernel reloads this with one v8f load.
            *(v8f*)(xz + (((long)gw * 16 + nt) * 256 + lane * 8)) = acc;
        }
    }
}

// ---------------------------------------------------------------------------
// Kernel 2: sequential LSTM scan. 32 blocks (one per 16-row batch tile),
// 4 waves/block. Wave w owns gate columns [w*16, w*16+16) of i/f/g/o
// (z tiles w, w+4, w+8, w+12) so the c/h update is wave-local in D layout.
// Wh B-fragments live in VGPRs for the whole scan; h goes through padded LDS.
// xz loads are software-pipelined one step ahead (independent of recursion),
// keeping the serial critical path = 8 WMMAs + elementwise + 2 barriers.
// ---------------------------------------------------------------------------
__global__ __launch_bounds__(128) void lstm_scan(const float* __restrict__ xz,
                                                 const _Float16* __restrict__ wht,
                                                 float* __restrict__ seq,
                                                 float* __restrict__ hT,
                                                 float* __restrict__ cT) {
    const int lane  = threadIdx.x & 31;
    const int w     = threadIdx.x >> 5;   // 0..3
    const int bt    = blockIdx.x;         // 0..31
    const int m     = lane & 15;
    const int khalf = (lane >> 4) * 8;
    const int HP    = 72;                 // padded LDS pitch (f16)

    __shared__ __align__(16) _Float16 hbuf[16 * 72];

    // Preload Wh B-fragments (4 N-tiles x 2 K-tiles x 8 VGPRs = 64 VGPRs).
    v16h bw[4][2];
#pragma unroll
    for (int j = 0; j < 4; ++j) {
        int nt = w + 4 * j;
#pragma unroll
        for (int kt = 0; kt < 2; ++kt) {
            const _Float16* bp = wht + (nt * 16 + m) * HDIM + kt * 32 + khalf;
            v8h lo = *(const v8h*)bp;
            v8h hi = *(const v8h*)(bp + 16);
            bw[j][kt] = combine16(lo, hi);
        }
    }

    // h_0 = 0
    for (int i = threadIdx.x; i < 16 * HP; i += blockDim.x) hbuf[i] = (_Float16)0.0f;
    __syncthreads();

    v8f c = { 0, 0, 0, 0, 0, 0, 0, 0 };
    float hv[8];
#pragma unroll
    for (int e = 0; e < 8; ++e) hv[e] = 0.0f;

    const float* xz_base = xz + (long)bt * 256 * 16 * 256;
    const int rbase = 8 * (lane >> 4);   // D-layout: row = rbase + acc element

    // Pipeline prologue: load z for t=0.
    v8f acc[4];
#pragma unroll
    for (int j = 0; j < 4; ++j)
        acc[j] = *(const v8f*)(xz_base + (((long)0 * 16 + (w + 4 * j)) * 256) + lane * 8);

    for (int t = 0; t < TLEN; ++t) {
        // Issue next step's xz loads early; they resolve during this step.
        v8f nxt[4];
        if (t + 1 < TLEN) {
#pragma unroll
            for (int j = 0; j < 4; ++j)
                nxt[j] = *(const v8f*)(xz_base + (((long)(t + 1) * 16 + (w + 4 * j)) * 256) + lane * 8);
        }
        if (t + 2 < TLEN) {  // hint t+2 into cache (global_prefetch_b8)
            __builtin_prefetch(xz_base + (((long)(t + 2) * 16 + w) * 256) + lane * 8, 0, 1);
        }

        // h A-fragments from LDS (2 K-tiles of 16x32 f16).
        v16h a[2];
#pragma unroll
        for (int kt = 0; kt < 2; ++kt) {
            const _Float16* ap = hbuf + m * HP + kt * 32 + khalf;
            v8h lo = *(const v8h*)ap;
            v8h hi = *(const v8h*)(ap + 16);
            a[kt] = combine16(lo, hi);
        }

        // z += h @ Wh : 8 WMMAs per wave per step.
#pragma unroll
        for (int j = 0; j < 4; ++j) {
#pragma unroll
            for (int kt = 0; kt < 2; ++kt) {
                acc[j] = __builtin_amdgcn_wmma_f32_16x16x32_f16(
                    false, a[kt], false, bw[j][kt], (short)0, acc[j], false, false);
            }
        }

        __syncthreads();  // all waves done reading hbuf before it is overwritten

        // Gates (Keras order i,f,g,o) -> c, h. All wave-local in D layout.
#pragma unroll
        for (int e = 0; e < 8; ++e) {
            float ig = sigmoidf_fast(acc[0][e]);
            float fg = sigmoidf_fast(acc[1][e]);
            float gg = tanhf_fast(acc[2][e]);
            float og = sigmoidf_fast(acc[3][e]);
            float cv = fg * c[e] + ig * gg;
            c[e] = cv;
            float h = og * tanhf_fast(cv);
            hv[e] = h;
            int r = rbase + e;
            hbuf[r * HP + w * 16 + m] = (_Float16)h;
            seq[(((long)(bt * 16 + r)) * TLEN + t) * HDIM + w * 16 + m] = h;
        }
        __syncthreads();  // h writes visible before next step's fragment reads

#pragma unroll
        for (int j = 0; j < 4; ++j) acc[j] = nxt[j];
    }

#pragma unroll
    for (int e = 0; e < 8; ++e) {
        int r = rbase + e;
        hT[(bt * 16 + r) * HDIM + w * 16 + m] = hv[e];
        cT[(bt * 16 + r) * HDIM + w * 16 + m] = c[e];
    }
}

extern "C" void kernel_launch(void* const* d_in, const int* in_sizes, int n_in,
                              void* d_out, int out_size, void* d_ws, size_t ws_size,
                              hipStream_t stream) {
    const int*   tokens = (const int*)d_in[0];
    const float* emb    = (const float*)d_in[1];
    const float* Wx     = (const float*)d_in[2];
    const float* Wh     = (const float*)d_in[3];
    const float* bias   = (const float*)d_in[4];
    float* out = (float*)d_out;

    _Float16* wxt = (_Float16*)d_ws;
    _Float16* wht = wxt + WS_WHT_ELEMS;
    float*    xz  = (float*)((char*)d_ws + WS_XZ_BYTES);

    float* seq = out;
    float* hT  = out + (long)BSZ * TLEN * HDIM;
    float* cT  = hT + (long)BSZ * HDIM;

    // 1) weight transpose/convert (tiny)
    prep_weights<<<(NH4 * DDIM + NH4 * HDIM + 255) / 256, 256, 0, stream>>>(Wx, Wh, wxt, wht);
    // 2) xz = emb[tokens] @ Wx + b  (8192 tiles, 8 per wave, WxT async-staged in LDS)
    xz_gemm<<<256, 128, 0, stream>>>(tokens, emb, wxt, bias, xz);
    // 3) sequential scan (32 independent batch tiles)
    lstm_scan<<<32, 128, 0, stream>>>(xz, wht, seq, hT, cT);
}
